// MambaDecoder_82145544503494
// MI455X (gfx1250) — matrix operand order repaired and checked by reading, not compiled, not run on hardware
//
#include <hip/hip_runtime.h>
#include <hip/hip_bf16.h>

// ---------------- model constants ----------------
#define BSZ     4
#define LSEQ    1024
#define NMELS   80
#define DMODEL  512
#define NLAYER  6
#define DSTATE  16
#define DINNER  1024
#define DTRANK  32
#define MROWS   (BSZ * LSEQ)   // 4096

typedef unsigned short u16;
typedef unsigned int   u32;
typedef u16   u16x8  __attribute__((ext_vector_type(8)));
typedef u16   u16x16 __attribute__((ext_vector_type(16)));
typedef u32   u32x4  __attribute__((ext_vector_type(4)));
typedef int   i32x4  __attribute__((ext_vector_type(4)));
typedef __bf16 v16bf __attribute__((ext_vector_type(16)));
typedef float v8f    __attribute__((ext_vector_type(8)));

__device__ inline u16 f2bf(float f) {
    unsigned int u = __float_as_uint(f);
    u += 0x7fffu + ((u >> 16) & 1u);   // RNE
    return (u16)(u >> 16);
}

__device__ inline v16bf cat16(u16x8 lo, u16x8 hi) {
    u16x16 c = __builtin_shufflevector(lo, hi, 0,1,2,3,4,5,6,7,8,9,10,11,12,13,14,15);
    return __builtin_bit_cast(v16bf, c);
}

__device__ inline v8f wmma_bf16(v16bf a, v16bf b, v8f c) {
    return __builtin_amdgcn_wmma_f32_16x16x32_bf16(false, a, false, b, (short)0, c,
                                                   false, false);
}

// ----- gfx1250 async global->LDS copy (fallback: synchronous staging) --------
#if __has_builtin(__builtin_amdgcn_global_load_async_to_lds_b128)
#define HAS_ASYNC_LDS 1
#else
#define HAS_ASYNC_LDS 0
#endif

typedef __attribute__((address_space(1))) i32x4 gas_i32x4;   // global (__device__)
typedef __attribute__((address_space(3))) i32x4 las_i32x4;   // LDS (__shared__)

__device__ inline void copy16_g2l(const u16* __restrict__ gsrc, u16* lds_dst) {
#if HAS_ASYNC_LDS
    __builtin_amdgcn_global_load_async_to_lds_b128(
        (gas_i32x4*)(u16*)gsrc,        // drop const, reinterpret to AS(1) i32x4*
        (las_i32x4*)lds_dst, 0, 0);
#else
    *(u32x4*)lds_dst = *(const u32x4*)gsrc;
#endif
}

__device__ inline void async_wait_2() {
#if HAS_ASYNC_LDS
#if __has_builtin(__builtin_amdgcn_s_wait_asynccnt)
    __builtin_amdgcn_s_wait_asynccnt(2);
#else
    asm volatile("s_wait_asynccnt 0x2" ::: "memory");
#endif
#endif
}
__device__ inline void async_wait_0() {
#if HAS_ASYNC_LDS
#if __has_builtin(__builtin_amdgcn_s_wait_asynccnt)
    __builtin_amdgcn_s_wait_asynccnt(0);
#else
    asm volatile("s_wait_asynccnt 0x0" ::: "memory");
#endif
#endif
}

// ---------------- weight prep: fp32 (N,K) -> bf16 transposed (Kpad, Npad) ----
__global__ __launch_bounds__(256)
void k_transpose_bf16(const float* __restrict__ src, u16* __restrict__ dst,
                      int N, int K, int Kpad, int Npad) {
    int idx = blockIdx.x * 256 + threadIdx.x;
    if (idx >= Kpad * Npad) return;
    int k = idx / Npad, n = idx % Npad;
    float v = (k < K && n < N) ? src[(size_t)n * K + k] : 0.0f;
    dst[idx] = f2bf(v);
}

// x (4096,80) fp32 -> (4096,96) bf16 zero-padded
__global__ __launch_bounds__(256)
void k_convert_x(const float* __restrict__ x, u16* __restrict__ x16) {
    int idx = blockIdx.x * 256 + threadIdx.x;
    if (idx >= MROWS * 96) return;
    int row = idx / 96, k = idx % 96;
    x16[idx] = (k < NMELS) ? f2bf(x[(size_t)row * NMELS + k]) : (u16)0;
}

// ---------------- bf16 WMMA GEMM with async-LDS B staging --------------------
// C(M,N) = A(M,K) * B(K,N) [+ epilogue].  Block = 256 thr = 8 waves (2 M x 4 N),
// block tile 128x128, wave tile 64x32.  B k-slab (32 x 128) double-buffered in
// LDS via GLOBAL_LOAD_ASYNC_TO_LDS_B128.  272-byte LDS row pitch -> the 32-lane
// 16-bit fragment reads are bank-conflict-free.
// mode: 0 = none, 1 = +bias[col], 2 = +resid[row,col], 3 = softplus(v+bias[col])
#define LDSB_STRIDE 136   // elements (= 272 bytes)

__global__ __launch_bounds__(256)
void k_gemm_bf16(const u16* __restrict__ A, int lda,
                 const u16* __restrict__ Bm, int ldb,
                 float* __restrict__ C, int ldc,
                 const float* __restrict__ bias,
                 const float* __restrict__ resid,
                 int M, int N, int K, int mode) {
    __shared__ u16 ldsB[2][32][LDSB_STRIDE];

    int t    = threadIdx.x;
    int lane = t & 31;
    int wave = t >> 5;
    int wm   = wave & 1;          // 2 waves along M
    int wn   = wave >> 1;         // 4 waves along N
    int m0   = blockIdx.x * 128 + wm * 64;
    int n0l  = wn * 32;                       // column offset inside block tile
    int n0   = blockIdx.y * 128 + n0l;
    int colBase = blockIdx.y * 128;
    bool active = (m0 + 64 <= M) && (n0 + 32 <= ldb);   // wave-uniform

    int lhi = lane >> 4, ln = lane & 15;

    v8f acc[4][2];
    #pragma unroll
    for (int mi = 0; mi < 4; ++mi)
        #pragma unroll
        for (int nj = 0; nj < 2; ++nj) acc[mi][nj] = (v8f){};

    // stage one 32xK-slab of B: 512 x 16B chunks, 2 per thread
    auto stage_B = [&](int kbase, int buf) {
        #pragma unroll
        for (int c = 0; c < 2; ++c) {
            int chunk = t + c * 256;
            int row = chunk >> 4, cidx = chunk & 15;
            int gcol = colBase + cidx * 8;
            if (gcol < ldb)
                copy16_g2l(Bm + (size_t)(kbase + row) * ldb + gcol,
                           &ldsB[buf][row][cidx * 8]);
        }
    };

    int T = K >> 5;
    stage_B(0, 0);
    for (int kt = 0; kt < T; ++kt) {
        if (kt + 1 < T) { stage_B((kt + 1) << 5, (kt + 1) & 1); async_wait_2(); }
        else            { async_wait_0(); }
        __syncthreads();

        if (active) {
            int kb = (kt << 5) + lhi * 8;
            // ---- A fragments from global (16-bit A 16x32 ISA layout) ----
            v16bf fa[4];
            #pragma unroll
            for (int mi = 0; mi < 4; ++mi) {
                const u16* ap = A + (size_t)(m0 + mi * 16 + ln) * lda + kb;
                fa[mi] = cat16(*(const u16x8*)ap, *(const u16x8*)(ap + 16));
            }
            // ---- B fragments from LDS: lane = N col, 16 K vals (table order) ----
            u16x16 b0v, b1v;
            #pragma unroll
            for (int s = 0; s < 16; ++s) {
                int kk = (s < 8) ? (lhi * 8 + s) : (16 + lhi * 8 + (s - 8));
                b0v[s] = ldsB[kt & 1][kk][n0l + ln];
                b1v[s] = ldsB[kt & 1][kk][n0l + 16 + ln];
            }
            v16bf fb0 = __builtin_bit_cast(v16bf, b0v);
            v16bf fb1 = __builtin_bit_cast(v16bf, b1v);
            #pragma unroll
            for (int mi = 0; mi < 4; ++mi) {
                acc[mi][0] = wmma_bf16(fa[mi], fb0, acc[mi][0]);
                acc[mi][1] = wmma_bf16(fa[mi], fb1, acc[mi][1]);
            }
        }
        __syncthreads();
    }

    if (!active) return;

    // --- epilogue: row = m0+mi*16+lhi*8+r, col = n0+nj*16+ln ---
    #pragma unroll
    for (int mi = 0; mi < 4; ++mi) {
        #pragma unroll
        for (int nj = 0; nj < 2; ++nj) {
            int rowBase = m0 + mi * 16 + lhi * 8;
            int col = n0 + nj * 16 + ln;
            #pragma unroll
            for (int r = 0; r < 8; ++r) {
                int row = rowBase + r;
                float v = acc[mi][nj][r];
                if (mode == 1)      v += bias[col];
                else if (mode == 2) v += resid[(size_t)row * ldc + col];
                else if (mode == 3) {
                    float xx = v + bias[col];
                    v = (xx > 20.0f) ? xx : logf(1.0f + __expf(xx));  // softplus
                }
                if (col < N) C[(size_t)row * ldc + col] = v;
            }
        }
    }
}

// ---------------- RMSNorm -> bf16 -------------------------------------------
__global__ __launch_bounds__(256)
void k_rmsnorm_bf16(const float* __restrict__ X, const float* __restrict__ w,
                    u16* __restrict__ out) {
    __shared__ float sred[256];
    int row = blockIdx.x, t = threadIdx.x;
    float v0 = X[(size_t)row * DMODEL + t];
    float v1 = X[(size_t)row * DMODEL + t + 256];
    sred[t] = v0 * v0 + v1 * v1;
    __syncthreads();
    for (int s = 128; s > 0; s >>= 1) {
        if (t < s) sred[t] += sred[t + s];
        __syncthreads();
    }
    float scale = rsqrtf(sred[0] * (1.0f / DMODEL) + 1e-5f);
    out[(size_t)row * DMODEL + t]       = f2bf(v0 * scale * w[t]);
    out[(size_t)row * DMODEL + t + 256] = f2bf(v1 * scale * w[t + 256]);
}

// ---------------- causal depthwise conv(4) + SiLU ----------------------------
__global__ __launch_bounds__(256)
void k_conv_silu(const float* __restrict__ xr, const float* __restrict__ cw,
                 const float* __restrict__ cb, float* __restrict__ uf,
                 u16* __restrict__ u16o) {
    int idx = blockIdx.x * 256 + threadIdx.x;           // (row, c)
    if (idx >= MROWS * DINNER) return;
    int row = idx >> 10, c = idx & 1023;
    int l = row & (LSEQ - 1);
    float acc = cb[c];
    #pragma unroll
    for (int j = 0; j < 4; ++j) {
        int lj = l - 3 + j;
        if (lj >= 0)
            acc += cw[c * 4 + j] * xr[(size_t)(row - 3 + j) * 2048 + c];
    }
    float s = acc / (1.0f + __expf(-acc));              // SiLU
    uf[idx] = s;
    u16o[idx] = f2bf(s);
}

// ---------------- extract dt (first 32 cols of xdbl) as bf16 -----------------
__global__ __launch_bounds__(256)
void k_dt_extract(const float* __restrict__ xdbl, u16* __restrict__ dt16) {
    int idx = blockIdx.x * 256 + threadIdx.x;
    if (idx >= MROWS * DTRANK) return;
    int row = idx >> 5, kk = idx & 31;
    dt16[idx] = f2bf(xdbl[(size_t)row * 64 + kk]);
}

// ---------------- selective scan --------------------------------------------
__global__ __launch_bounds__(256)
void k_scan(const float* __restrict__ delta, const float* __restrict__ uf,
            const float* __restrict__ xdbl,  const float* __restrict__ xr,
            const float* __restrict__ A_log, const float* __restrict__ Dp,
            u16* __restrict__ y16) {
    int lane = threadIdx.x & 31;
    int wave = blockIdx.x * 8 + (threadIdx.x >> 5);
    int half = lane >> 4, n = lane & 15;
    int g = wave * 2 + half;                 // [0, 4096) = b*DINNER + d
    int b = g >> 10, d = g & 1023;
    float Aval = -__expf(A_log[d * DSTATE + n]);
    float Dval = Dp[d];
    float h = 0.0f;
    int rowBase = b << 10;
    for (int l = 0; l < LSEQ; ++l) {
        int row = rowBase + l;
        float dl = delta[(size_t)row * DINNER + d];
        float uu = uf[(size_t)row * DINNER + d];
        float Bn = xdbl[(size_t)row * 64 + 32 + n];
        float Cn = xdbl[(size_t)row * 64 + 48 + n];
        float dA = __expf(dl * Aval);
        h = dA * h + dl * Bn * uu;
        float part = h * Cn;
        part += __shfl_xor(part, 1);
        part += __shfl_xor(part, 2);
        part += __shfl_xor(part, 4);
        part += __shfl_xor(part, 8);        // sum over 16 state lanes
        if (n == 0) {
            float y = part + uu * Dval;
            float rv = xr[(size_t)row * 2048 + 1024 + d];
            y *= rv / (1.0f + __expf(-rv)); // * SiLU(res)
            y16[(size_t)row * DINNER + d] = f2bf(y);
        }
    }
}

// ---------------- host orchestration ----------------------------------------
static inline size_t alignup(size_t x) { return (x + 255) & ~(size_t)255; }

extern "C" void kernel_launch(void* const* d_in, const int* in_sizes, int n_in,
                              void* d_out, int out_size, void* d_ws, size_t ws_size,
                              hipStream_t stream) {
    const float* x        = (const float*)d_in[0];
    const float* in_w     = (const float*)d_in[1];
    const float* in_b     = (const float*)d_in[2];
    const float* norm_w   = (const float*)d_in[3];
    const float* inproj_w = (const float*)d_in[4];
    const float* conv_w   = (const float*)d_in[5];
    const float* conv_b   = (const float*)d_in[6];
    const float* xproj_w  = (const float*)d_in[7];
    const float* dt_w     = (const float*)d_in[8];
    const float* dt_b     = (const float*)d_in[9];
    const float* A_log    = (const float*)d_in[10];
    const float* Dp       = (const float*)d_in[11];
    const float* outproj_w= (const float*)d_in[12];
    const float* normf_w  = (const float*)d_in[13];
    const float* out_w    = (const float*)d_in[14];

    char* base = (char*)d_ws;
    size_t off = 0;
    auto carve = [&](size_t bytes) { char* p = base + off; off += alignup(bytes); return p; };

    float* ws_h      = (float*)carve((size_t)MROWS * DMODEL * 4);
    u16*   ws_normed = (u16*)  carve((size_t)MROWS * DMODEL * 2);
    float* ws_xr     = (float*)carve((size_t)MROWS * 2 * DINNER * 4);
    float* ws_uf     = (float*)carve((size_t)MROWS * DINNER * 4);
    u16*   ws_u16    = (u16*)  carve((size_t)MROWS * DINNER * 2);
    float* ws_xdbl   = (float*)carve((size_t)MROWS * 64 * 4);
    u16*   ws_dt16   = (u16*)  carve((size_t)MROWS * DTRANK * 2);
    float* ws_delta  = (float*)carve((size_t)MROWS * DINNER * 4);
    u16*   ws_y16    = (u16*)  carve((size_t)MROWS * DINNER * 2);
    u16*   ws_x16    = (u16*)  carve((size_t)MROWS * 96 * 2);
    u16*   w_in      = (u16*)  carve((size_t)96 * DMODEL * 2);
    u16*   w_inproj  = (u16*)  carve((size_t)NLAYER * DMODEL * 2 * DINNER * 2);
    u16*   w_xproj   = (u16*)  carve((size_t)NLAYER * DINNER * 64 * 2);
    u16*   w_dt      = (u16*)  carve((size_t)NLAYER * DTRANK * DINNER * 2);
    u16*   w_outproj = (u16*)  carve((size_t)NLAYER * DINNER * DMODEL * 2);
    u16*   w_out     = (u16*)  carve((size_t)DMODEL * 96 * 2);

    auto tgrid = [](int n) { return dim3((n + 255) / 256); };

    // ---- weight prep (fp32 -> bf16, transposed to K-major) ----
    k_transpose_bf16<<<tgrid(96 * DMODEL), 256, 0, stream>>>(
        in_w, w_in, DMODEL, NMELS, 96, DMODEL);
    for (int i = 0; i < NLAYER; ++i) {
        k_transpose_bf16<<<tgrid(DMODEL * 2 * DINNER), 256, 0, stream>>>(
            inproj_w + (size_t)i * 2 * DINNER * DMODEL,
            w_inproj + (size_t)i * DMODEL * 2 * DINNER,
            2 * DINNER, DMODEL, DMODEL, 2 * DINNER);
        k_transpose_bf16<<<tgrid(DINNER * 64), 256, 0, stream>>>(
            xproj_w + (size_t)i * 64 * DINNER,
            w_xproj + (size_t)i * DINNER * 64,
            64, DINNER, DINNER, 64);
        k_transpose_bf16<<<tgrid(DTRANK * DINNER), 256, 0, stream>>>(
            dt_w + (size_t)i * DINNER * DTRANK,
            w_dt + (size_t)i * DTRANK * DINNER,
            DINNER, DTRANK, DTRANK, DINNER);
        k_transpose_bf16<<<tgrid(DINNER * DMODEL), 256, 0, stream>>>(
            outproj_w + (size_t)i * DMODEL * DINNER,
            w_outproj + (size_t)i * DINNER * DMODEL,
            DMODEL, DINNER, DINNER, DMODEL);
    }
    k_transpose_bf16<<<tgrid(DMODEL * 96), 256, 0, stream>>>(
        out_w, w_out, NMELS, DMODEL, DMODEL, 96);
    k_convert_x<<<tgrid(MROWS * 96), 256, 0, stream>>>(x, ws_x16);

    // ---- input projection: h = x @ in_w^T + in_b ----
    k_gemm_bf16<<<dim3(MROWS / 128, 4), 256, 0, stream>>>(
        ws_x16, 96, w_in, DMODEL, ws_h, DMODEL, in_b, nullptr,
        MROWS, DMODEL, 96, 1);

    // ---- layers ----
    for (int i = 0; i < NLAYER; ++i) {
        k_rmsnorm_bf16<<<MROWS, 256, 0, stream>>>(ws_h, norm_w + (size_t)i * DMODEL,
                                                  ws_normed);
        // xr = normed @ inproj^T  (4096 x 2048)
        k_gemm_bf16<<<dim3(MROWS / 128, 16), 256, 0, stream>>>(
            ws_normed, DMODEL, w_inproj + (size_t)i * DMODEL * 2 * DINNER, 2 * DINNER,
            ws_xr, 2 * DINNER, nullptr, nullptr, MROWS, 2 * DINNER, DMODEL, 0);
        // u = SiLU(causal depthwise conv(u) + b)
        k_conv_silu<<<tgrid(MROWS * DINNER), 256, 0, stream>>>(
            ws_xr, conv_w + (size_t)i * DINNER * 4, conv_b + (size_t)i * DINNER,
            ws_uf, ws_u16);
        // xdbl = u @ xproj^T  (4096 x 64)
        k_gemm_bf16<<<dim3(MROWS / 128, 1), 256, 0, stream>>>(
            ws_u16, DINNER, w_xproj + (size_t)i * DINNER * 64, 64,
            ws_xdbl, 64, nullptr, nullptr, MROWS, 64, DINNER, 0);
        k_dt_extract<<<tgrid(MROWS * DTRANK), 256, 0, stream>>>(ws_xdbl, ws_dt16);
        // delta = softplus(dt @ dt_w^T + dt_b)  (4096 x 1024)
        k_gemm_bf16<<<dim3(MROWS / 128, 8), 256, 0, stream>>>(
            ws_dt16, DTRANK, w_dt + (size_t)i * DTRANK * DINNER, DINNER,
            ws_delta, DINNER, dt_b + (size_t)i * DINNER, nullptr,
            MROWS, DINNER, DTRANK, 3);
        // selective scan + gating -> y16 (bf16)
        k_scan<<<MROWS / 16, 256, 0, stream>>>(
            ws_delta, ws_uf, ws_xdbl, ws_xr,
            A_log + (size_t)i * DINNER * DSTATE, Dp + (size_t)i * DINNER, ws_y16);
        // h += y @ outproj^T  (residual fused in epilogue, in-place)
        k_gemm_bf16<<<dim3(MROWS / 128, 4), 256, 0, stream>>>(
            ws_y16, DINNER, w_outproj + (size_t)i * DINNER * DMODEL, DMODEL,
            ws_h, DMODEL, nullptr, ws_h, MROWS, DMODEL, DINNER, 2);
    }

    // ---- final norm + output projection ----
    k_rmsnorm_bf16<<<MROWS, 256, 0, stream>>>(ws_h, normf_w, ws_normed);
    k_gemm_bf16<<<dim3(MROWS / 128, 1), 256, 0, stream>>>(
        ws_normed, DMODEL, w_out, 96, (float*)d_out, NMELS, nullptr, nullptr,
        MROWS, NMELS, DMODEL, 0);
}
